// HypergraphConv_11158325035087
// MI455X (gfx1250) — matrix operand order repaired and checked by reading, not compiled, not run on hardware
//
#include <hip/hip_runtime.h>
#include <hip/hip_bf16.h>

// HypergraphConv on MI455X (gfx1250, wave32, WMMA).
//
// Pipeline (all bf16 WMMA, f32 accumulate):
//  K0: W_node/W_res -> transposed bf16 [512,256]
//  K1: H f32 [N,M] -> H_bf16 [N,M_PAD] and HT_bf16 [M_PAD,N] (one 400MB read, all NT)
//  K2: P_T[f,n] = (x @ W_node)^T  bf16           (P_T stays L2-resident)
//  K3: E_T[f,m] = ((H^T @ P)/d_e)^T              (A streamed NT; d_e fused from A sums)
//  K4: out = LayerNorm((H @ E)/d_v + x @ W_res)  (A streamed NT; d_v + LN fused)

#define N_NODES 20000
#define M_EDGES 5000
#define M_PAD   5024     // multiple of 32; pad rows/cols are zero
#define D_IN    256
#define D_OUT   512
#define EPS_LN  1e-5f

typedef __attribute__((ext_vector_type(16))) __bf16 v16bf;
typedef __attribute__((ext_vector_type(8)))  float  v8f;
typedef __attribute__((ext_vector_type(4)))  unsigned int v4u;

union V16BF { v16bf v; __bf16 e[16]; v4u u[2]; };
union PACK8 { __bf16 e[8]; v4u u; };
union BFS   { __bf16 b; unsigned short s; };

// A-matrix 16x32 bf16 per ISA: lane lr holds row M=lr; VGPR0..3 = K (k0+8*hi .. +7),
// VGPR4..7 = K (k0+16+8*hi .. +7). Two 16B contiguous loads per lane.
// Non-temporal: A (the H stream) is touch-once; keep it out of L2 so B stays resident.
static __device__ inline V16BF load_a_bf16_nt(const __bf16* __restrict__ base, int lda,
                                              int row, int k0, int hi) {
  V16BF r;
  const __bf16* p = base + (size_t)row * lda + k0 + 8 * hi;
  r.u[0] = __builtin_nontemporal_load((const v4u*)p);
  r.u[1] = __builtin_nontemporal_load((const v4u*)(p + 16));
  return r;
}

// B-matrix 32x16 bf16: lane lr holds column N=lr; K (k0+8*hi .. +7) then K (k0+16+8*hi .. +7).
// B is stored K-contiguous (transposed layout [col][K]) -> two 16B contiguous loads.
// Regular temporal policy: B matrices are small and hot (L2-resident).
static __device__ inline v16bf load_b_bf16(const __bf16* __restrict__ base, int ldb,
                                           int col, int k0, int hi) {
  V16BF r;
  const __bf16* p = base + (size_t)col * ldb + k0 + 8 * hi;
  r.u[0] = *(const v4u*)p;
  r.u[1] = *(const v4u*)(p + 16);
  return r.v;
}

// A-tile loaded from f32 source (x), converted to bf16 on the fly.
static __device__ inline V16BF load_a_f32(const float* __restrict__ base, int lda,
                                          int row, int k0, int hi) {
  V16BF r;
  const float* p = base + (size_t)row * lda + k0 + 8 * hi;
  float4 f0 = *(const float4*)(p);
  float4 f1 = *(const float4*)(p + 4);
  float4 f2 = *(const float4*)(p + 16);
  float4 f3 = *(const float4*)(p + 20);
  r.e[0]  = (__bf16)f0.x; r.e[1]  = (__bf16)f0.y; r.e[2]  = (__bf16)f0.z; r.e[3]  = (__bf16)f0.w;
  r.e[4]  = (__bf16)f1.x; r.e[5]  = (__bf16)f1.y; r.e[6]  = (__bf16)f1.z; r.e[7]  = (__bf16)f1.w;
  r.e[8]  = (__bf16)f2.x; r.e[9]  = (__bf16)f2.y; r.e[10] = (__bf16)f2.z; r.e[11] = (__bf16)f2.w;
  r.e[12] = (__bf16)f3.x; r.e[13] = (__bf16)f3.y; r.e[14] = (__bf16)f3.z; r.e[15] = (__bf16)f3.w;
  return r;
}

// ---------------- K0: transpose+convert weight matrices ----------------
__global__ void k_prep_w(const float* __restrict__ Wn, const float* __restrict__ Wr,
                         __bf16* __restrict__ WnT, __bf16* __restrict__ WrT) {
  int idx = blockIdx.x * 256 + threadIdx.x;
  if (idx < D_OUT * D_IN) {
    int f = idx / D_IN, k = idx % D_IN;
    WnT[idx] = (__bf16)Wn[(size_t)k * D_OUT + f];
    WrT[idx] = (__bf16)Wr[(size_t)k * D_OUT + f];
  }
}

// ---------------- K1: H f32 -> H_bf16 and HT_bf16 (LDS transpose, all NT) ----------------
__global__ void k_convH(const float* __restrict__ H, __bf16* __restrict__ Hb,
                        __bf16* __restrict__ HTb) {
  __shared__ __bf16 tile[32][33];
  int n0 = blockIdx.x * 32;
  int m0 = blockIdx.y * 32;
  int tx = threadIdx.x & 31;
  int ty = threadIdx.x >> 5;  // 0..7
#pragma unroll
  for (int i = 0; i < 4; ++i) {
    int r = ty + 8 * i;  // local n
    int m = m0 + tx;
    float v = (m < M_EDGES)
                  ? __builtin_nontemporal_load(&H[(size_t)(n0 + r) * M_EDGES + m])
                  : 0.0f;
    BFS b; b.b = (__bf16)v;
    tile[r][tx] = b.b;
    __builtin_nontemporal_store(
        b.s, (unsigned short*)&Hb[(size_t)(n0 + r) * M_PAD + m0 + tx]);
  }
  __syncthreads();
#pragma unroll
  for (int i = 0; i < 4; ++i) {
    int r = ty + 8 * i;  // local m
    BFS b; b.b = tile[tx][r];
    __builtin_nontemporal_store(
        b.s, (unsigned short*)&HTb[(size_t)(m0 + r) * N_NODES + n0 + tx]);
  }
}

// ---------------- K2: P_T = (x @ W_node)^T, bf16 ----------------
__global__ void __launch_bounds__(256) k_xproj(const float* __restrict__ x,
                                               const __bf16* __restrict__ WnT,
                                               __bf16* __restrict__ PT) {
  int n0 = blockIdx.x * 16;
  int wave = threadIdx.x >> 5, lane = threadIdx.x & 31;
  int hi = lane >> 4, lr = lane & 15;
  v8f acc[4] = {};
  for (int k0 = 0; k0 < D_IN; k0 += 32) {
    V16BF a = load_a_f32(x, D_IN, n0 + lr, k0, hi);
#pragma unroll
    for (int t = 0; t < 4; ++t) {
      int f = (wave * 4 + t) * 16 + lr;
      v16bf b = load_b_bf16(WnT, D_IN, f, k0, hi);
      acc[t] = __builtin_amdgcn_wmma_f32_16x16x32_bf16(false, a.v, false, b,
                                                       (short)0, acc[t], false, false);
    }
  }
#pragma unroll
  for (int t = 0; t < 4; ++t) {
    int f = (wave * 4 + t) * 16 + lr;
    PACK8 pk;
#pragma unroll
    for (int v = 0; v < 8; ++v) pk.e[v] = (__bf16)acc[t][v];
    // D element (row n0+v+8*hi, col f) -> PT[f][n]: 8 consecutive n, one 16B store
    *(v4u*)(PT + (size_t)f * N_NODES + n0 + 8 * hi) = pk.u;
  }
}

// ---------------- K3: E_T = ((H^T @ P) / d_e)^T, d_e fused ----------------
__global__ void __launch_bounds__(256) k_edge(const __bf16* __restrict__ HTb,
                                              const __bf16* __restrict__ PT,
                                              __bf16* __restrict__ ET) {
  int m0 = blockIdx.x * 16;
  int wave = threadIdx.x >> 5, lane = threadIdx.x & 31;
  int hi = lane >> 4, lr = lane & 15;
  v8f acc[4] = {};
  float ds = 0.0f;  // partial degree of row (m0+lr)
  for (int k0 = 0; k0 < N_NODES; k0 += 32) {
    V16BF a = load_a_bf16_nt(HTb, N_NODES, m0 + lr, k0, hi);
#pragma unroll
    for (int i = 0; i < 16; ++i) ds += (float)a.e[i];
#pragma unroll
    for (int t = 0; t < 4; ++t) {
      int f = (wave * 4 + t) * 16 + lr;
      v16bf b = load_b_bf16(PT, N_NODES, f, k0, hi);
      acc[t] = __builtin_amdgcn_wmma_f32_16x16x32_bf16(false, a.v, false, b,
                                                       (short)0, acc[t], false, false);
    }
  }
  ds += __shfl_xor(ds, 16, 32);  // lanes lr and lr+16 each hold half the K range
  float sc[8];
#pragma unroll
  for (int v = 0; v < 8; ++v) {  // D rows are v+8*hi; their degree lives in lane (v+8*hi)
    float dsum = __shfl(ds, v + 8 * hi, 32);
    sc[v] = 1.0f / fmaxf(dsum, 1.0f);
  }
#pragma unroll
  for (int t = 0; t < 4; ++t) {
    int f = (wave * 4 + t) * 16 + lr;
    PACK8 pk;
#pragma unroll
    for (int v = 0; v < 8; ++v) pk.e[v] = (__bf16)(acc[t][v] * sc[v]);
    *(v4u*)(ET + (size_t)f * M_PAD + m0 + 8 * hi) = pk.u;
  }
}

// ---------------- K4: out = LN((H @ E)/d_v + x @ W_res) ----------------
__global__ void __launch_bounds__(256) k_node(const __bf16* __restrict__ Hb,
                                              const __bf16* __restrict__ ET,
                                              const float* __restrict__ x,
                                              const __bf16* __restrict__ WrT,
                                              const float* __restrict__ gamma,
                                              const float* __restrict__ beta,
                                              float* __restrict__ out) {
  __shared__ float hbuf[16][D_OUT + 8];
  __shared__ float s_mu[16], s_rs[16];
  int n0 = blockIdx.x * 16;
  int wave = threadIdx.x >> 5, lane = threadIdx.x & 31;
  int hi = lane >> 4, lr = lane & 15;
  v8f acc[4] = {};
  float ds = 0.0f;  // node degree of row (n0+lr)
  for (int k0 = 0; k0 < M_PAD; k0 += 32) {
    V16BF a = load_a_bf16_nt(Hb, M_PAD, n0 + lr, k0, hi);
#pragma unroll
    for (int i = 0; i < 16; ++i) ds += (float)a.e[i];
#pragma unroll
    for (int t = 0; t < 4; ++t) {
      int f = (wave * 4 + t) * 16 + lr;
      v16bf b = load_b_bf16(ET, M_PAD, f, k0, hi);
      acc[t] = __builtin_amdgcn_wmma_f32_16x16x32_bf16(false, a.v, false, b,
                                                       (short)0, acc[t], false, false);
    }
  }
  ds += __shfl_xor(ds, 16, 32);
  float sc[8];
#pragma unroll
  for (int v = 0; v < 8; ++v) {
    float dsum = __shfl(ds, v + 8 * hi, 32);
    sc[v] = 1.0f / fmaxf(dsum, 1.0f);
  }
#pragma unroll
  for (int t = 0; t < 4; ++t)
#pragma unroll
    for (int v = 0; v < 8; ++v) acc[t][v] *= sc[v];
  // fused residual: += x @ W_res (scaled acc feeds the WMMA C input)
  for (int k0 = 0; k0 < D_IN; k0 += 32) {
    V16BF a = load_a_f32(x, D_IN, n0 + lr, k0, hi);
#pragma unroll
    for (int t = 0; t < 4; ++t) {
      int f = (wave * 4 + t) * 16 + lr;
      v16bf b = load_b_bf16(WrT, D_IN, f, k0, hi);
      acc[t] = __builtin_amdgcn_wmma_f32_16x16x32_bf16(false, a.v, false, b,
                                                       (short)0, acc[t], false, false);
    }
  }
  // spill D tiles to LDS: element (row v+8*hi, col f)
#pragma unroll
  for (int t = 0; t < 4; ++t) {
    int f = (wave * 4 + t) * 16 + lr;
#pragma unroll
    for (int v = 0; v < 8; ++v) hbuf[v + 8 * hi][f] = acc[t][v];
  }
  __syncthreads();
  // LayerNorm: 16 threads per row
  int r = threadIdx.x >> 4;
  int c = threadIdx.x & 15;
  float s = 0.0f, sq = 0.0f;
  for (int f = c; f < D_OUT; f += 16) {
    float h = hbuf[r][f];
    s += h; sq += h * h;
  }
#pragma unroll
  for (int o = 8; o >= 1; o >>= 1) {  // xor masks <16 stay inside the 16-lane group
    s += __shfl_xor(s, o, 32);
    sq += __shfl_xor(sq, o, 32);
  }
  if (c == 0) {
    float mu = s * (1.0f / D_OUT);
    float var = sq * (1.0f / D_OUT) - mu * mu;
    s_mu[r] = mu;
    s_rs[r] = rsqrtf(fmaxf(var, 0.0f) + EPS_LN);
  }
  __syncthreads();
#pragma unroll
  for (int i = 0; i < 32; ++i) {
    int e = threadIdx.x + 256 * i;  // 16*512 elements
    int rr = e >> 9, f = e & (D_OUT - 1);
    out[(size_t)(n0 + rr) * D_OUT + f] =
        (hbuf[rr][f] - s_mu[rr]) * s_rs[rr] * gamma[f] + beta[f];
  }
}

extern "C" void kernel_launch(void* const* d_in, const int* in_sizes, int n_in,
                              void* d_out, int out_size, void* d_ws, size_t ws_size,
                              hipStream_t stream) {
  (void)in_sizes; (void)n_in; (void)out_size; (void)ws_size;
  const float* x     = (const float*)d_in[0];
  const float* H     = (const float*)d_in[1];
  const float* Wn    = (const float*)d_in[2];
  const float* Wr    = (const float*)d_in[3];
  const float* gamma = (const float*)d_in[4];
  const float* beta  = (const float*)d_in[5];
  float* out = (float*)d_out;

  char* ws = (char*)d_ws;
  size_t off = 0;
  auto alloc = [&](size_t bytes) -> void* {
    void* p = ws + off;
    off += (bytes + 255) & ~(size_t)255;
    return p;
  };
  __bf16* HTb = (__bf16*)alloc((size_t)M_PAD * N_NODES * 2);  // 201 MB
  __bf16* Hb  = (__bf16*)alloc((size_t)N_NODES * M_PAD * 2);  // 201 MB
  __bf16* PT  = (__bf16*)alloc((size_t)D_OUT * N_NODES * 2);  // 20.5 MB
  __bf16* ET  = (__bf16*)alloc((size_t)D_OUT * M_PAD * 2);    // 5.1 MB
  __bf16* WnT = (__bf16*)alloc((size_t)D_OUT * D_IN * 2);
  __bf16* WrT = (__bf16*)alloc((size_t)D_OUT * D_IN * 2);

  k_prep_w<<<(D_OUT * D_IN + 255) / 256, 256, 0, stream>>>(Wn, Wr, WnT, WrT);
  k_convH<<<dim3(N_NODES / 32, M_PAD / 32), 256, 0, stream>>>(H, Hb, HTb);
  k_xproj<<<N_NODES / 16, 256, 0, stream>>>(x, WnT, PT);
  k_edge<<<M_PAD / 16, 256, 0, stream>>>(HTb, PT, ET);
  k_node<<<N_NODES / 16, 256, 0, stream>>>(Hb, ET, x, WrT, gamma, beta, out);
}
